// ComplexConv2D_45896020525368
// MI455X (gfx1250) — matrix operand order repaired and checked by reading, not compiled, not run on hardware
//
#include <hip/hip_runtime.h>

// ---------------------------------------------------------------------------
// Complex 3x3 conv (NHWC, SAME) via implicit GEMM on CDNA5 WMMA.
//   real = conv(xr,kr) - conv(xi,ki) + br
//   imag = conv(xr,ki) + conv(xi,kr) + bi
// fp32 inputs -> bf16 hi/lo split (3-product emulation) -> v_wmma_f32_16x16x32_bf16
// ---------------------------------------------------------------------------

typedef __attribute__((ext_vector_type(16))) __bf16          v16bf;
typedef __attribute__((ext_vector_type(8)))  __bf16          v8bf;
typedef __attribute__((ext_vector_type(8)))  float           v8f;
typedef __attribute__((ext_vector_type(16))) unsigned short  v16u;

#define CB   16
#define CH   128
#define CW   128
#define CIN  64
#define CF   128

// ws layout (bf16 elements): [s(18)][p(2)][t(2)][nt(8)][lane(32)][j(16)]
//   s = (ky*3+kx)*2 + cc  (cc = cin chunk of 32)
//   value = w[ky][kx][cc*32 + (j<8 ? (lane>>4)*8 + j : 16 + (lane>>4)*8 + j-8)][nt*16 + (lane&15)]  (hi or lo part)
#define WS_T_STRIDE   4096      // 8*32*16
#define WS_P_STRIDE   8192
#define WS_S_STRIDE   16384
#define WS_TOTAL      (18 * WS_S_STRIDE)   // 294912 bf16 = 576 KB

// LDS layout (bf16): [p(2)][t(2)][row(3)][col(34)][slot(72: 64 cin + pad)]
#define XS_CSTRIDE    72
#define XS_ROWSTRIDE  (34 * XS_CSTRIDE)    // 2448
#define XS_T_STRIDE   (3 * XS_ROWSTRIDE)   // 7344
#define XS_P_STRIDE   (2 * XS_T_STRIDE)    // 14688
#define XS_TOTAL      (2 * XS_P_STRIDE)    // 29376 elems = 58752 B

static __device__ inline v8f wmma_bf16(v16bf a, v16bf b, v8f c) {
  return __builtin_amdgcn_wmma_f32_16x16x32_bf16(
      /*neg_a=*/false, a, /*neg_b=*/false, b,
      /*c_mod=*/(short)0, c, /*reuse_a=*/false, /*reuse_b=*/false);
}

// A fragment: 16 bf16 per lane from two contiguous 16-byte LDS chunks
static __device__ inline v16bf ld_a(const __bf16* p) {
  v8bf lo = *(const v8bf*)(p);
  v8bf hi = *(const v8bf*)(p + 16);
  v16bf r;
#pragma unroll
  for (int j = 0; j < 8; ++j) { r[j] = lo[j]; r[8 + j] = hi[j]; }
  return r;
}

// B fragment: 16 contiguous bf16 (32B aligned) from packed weights
static __device__ inline v16bf ld_b(const __bf16* p) {
  return *(const v16bf*)(p);
}

static __device__ inline v16bf neg16(v16bf x) {   // exact sign flip
  v16u u = __builtin_bit_cast(v16u, x);
  u = u ^ (v16u)0x8000;
  return __builtin_bit_cast(v16bf, u);
}

// ---------------------------------------------------------------------------
// Kernel 1: pack fp32 weights -> bf16 hi/lo fragments in ws
// ---------------------------------------------------------------------------
__global__ __launch_bounds__(256) void pack_weights_kernel(
    const float* __restrict__ kr, const float* __restrict__ ki,
    unsigned short* __restrict__ ws_u) {
  __bf16* ws = (__bf16*)ws_u;
  int idx = blockIdx.x * 256 + threadIdx.x;
  if (idx >= WS_TOTAL) return;
  int j  = idx & 15;
  int l  = (idx >> 4) & 31;
  int nt = (idx >> 9) & 7;
  int t  = (idx >> 12) & 1;
  int p  = (idx >> 13) & 1;
  int s  = idx >> 14;               // 0..17
  int cc = s & 1;
  int kk = s >> 1;                  // 0..8
  int cin = cc * 32 + ((j < 8) ? ((l >> 4) * 8 + j) : (16 + (l >> 4) * 8 + (j - 8)));
  int f   = nt * 16 + (l & 15);
  const float* src = p ? ki : kr;
  float w = src[(kk * CIN + cin) * CF + f];
  __bf16 hi = (__bf16)w;
  if (t == 0) {
    ws[idx] = hi;
  } else {
    float rem = w - (float)hi;
    ws[idx] = (__bf16)rem;
  }
}

// ---------------------------------------------------------------------------
// Kernel 2: main conv. Block = 256 threads (8 waves), tile = 32 pix x 128 F.
// ---------------------------------------------------------------------------
__global__ __launch_bounds__(256) void complex_conv_wmma_kernel(
    const float* __restrict__ xr, const float* __restrict__ xi,
    const float* __restrict__ br, const float* __restrict__ bi,
    const unsigned short* __restrict__ ws_u, float* __restrict__ out) {
  const __bf16* wsb = (const __bf16*)ws_u;
  __shared__ __bf16 xs[XS_TOTAL];

  const int tid = threadIdx.x;
  const int w0  = blockIdx.x * 32;
  const int h   = blockIdx.y;
  const int b   = blockIdx.z;

  // ---- cooperative stage of 3x34x64 halo (xr,xi) as bf16 hi/lo ----
#pragma unroll 1
  for (int i = tid; i < 2 * 3 * 34 * 64; i += 256) {
    int c     = i & 63;
    int rest  = i >> 6;        // 0..203
    int col   = rest % 34;
    int rest2 = rest / 34;     // 0..5
    int r     = rest2 % 3;
    int p     = rest2 / 3;     // 0..1
    int gh = h + r - 1;
    int gw = w0 + col - 1;
    float v = 0.0f;
    if ((unsigned)gh < (unsigned)CH && (unsigned)gw < (unsigned)CW) {
      const float* src = p ? xi : xr;
      v = src[(((size_t)b * CH + gh) * CW + gw) * CIN + c];
    }
    __bf16 hi = (__bf16)v;
    __bf16 lo = (__bf16)(v - (float)hi);
    int base = (p * XS_P_STRIDE) + (r * XS_ROWSTRIDE) + col * XS_CSTRIDE + c;
    xs[base]               = hi;
    xs[base + XS_T_STRIDE] = lo;
  }
  __syncthreads();   // one barrier; LDS tile covers the whole reduction

  const int lane = tid & 31;
  const int wave = tid >> 5;
  const int m0   = (wave & 1) * 16;   // pixel sub-tile
  const int ng   = wave >> 1;         // 0..3 -> 32 F columns
  const int nt0  = ng * 2;
  const int nt1  = nt0 + 1;
  const int mrow = lane & 15;
  const int half = lane >> 4;

  v8f accR0 = {}, accI0 = {}, accR1 = {}, accI1 = {};

#pragma unroll 1
  for (int kk = 0; kk < 9; ++kk) {
    const int ky   = kk / 3;
    const int kx   = kk - ky * 3;
    const int colL = m0 + mrow + kx;            // 0..33
#pragma unroll
    for (int cc = 0; cc < 2; ++cc) {
      const int s = kk * 2 + cc;
      const __bf16* wbase = wsb + (size_t)s * WS_S_STRIDE + (size_t)lane * 16;

      // --- B fragments for this reduction step (kr, ki) x (hi, lo) x 2 N-tiles
      v16bf krh0 = ld_b(wbase + nt0 * 512);
      v16bf krl0 = ld_b(wbase + nt0 * 512 + WS_T_STRIDE);
      v16bf krh1 = ld_b(wbase + nt1 * 512);
      v16bf krl1 = ld_b(wbase + nt1 * 512 + WS_T_STRIDE);
      v16bf kih0 = ld_b(wbase + WS_P_STRIDE + nt0 * 512);
      v16bf kil0 = ld_b(wbase + WS_P_STRIDE + nt0 * 512 + WS_T_STRIDE);
      v16bf kih1 = ld_b(wbase + WS_P_STRIDE + nt1 * 512);
      v16bf kil1 = ld_b(wbase + WS_P_STRIDE + nt1 * 512 + WS_T_STRIDE);

      const int abase = (ky * XS_ROWSTRIDE) + colL * XS_CSTRIDE + cc * 32 + half * 8;

      // ---- part xr : real += xr*kr ; imag += xr*ki ----
      {
        v16bf ah = ld_a(&xs[abase]);
        v16bf al = ld_a(&xs[abase + XS_T_STRIDE]);
        accR0 = wmma_bf16(ah, krh0, accR0);
        accR0 = wmma_bf16(ah, krl0, accR0);
        accR0 = wmma_bf16(al, krh0, accR0);
        accI0 = wmma_bf16(ah, kih0, accI0);
        accI0 = wmma_bf16(ah, kil0, accI0);
        accI0 = wmma_bf16(al, kih0, accI0);
        accR1 = wmma_bf16(ah, krh1, accR1);
        accR1 = wmma_bf16(ah, krl1, accR1);
        accR1 = wmma_bf16(al, krh1, accR1);
        accI1 = wmma_bf16(ah, kih1, accI1);
        accI1 = wmma_bf16(ah, kil1, accI1);
        accI1 = wmma_bf16(al, kih1, accI1);
      }
      // ---- part xi : real -= xi*ki ; imag += xi*kr ----
      {
        v16bf ah = ld_a(&xs[abase + XS_P_STRIDE]);
        v16bf al = ld_a(&xs[abase + XS_P_STRIDE + XS_T_STRIDE]);
        v16bf nh0 = neg16(kih0), nl0 = neg16(kil0);
        v16bf nh1 = neg16(kih1), nl1 = neg16(kil1);
        accR0 = wmma_bf16(ah, nh0, accR0);
        accR0 = wmma_bf16(ah, nl0, accR0);
        accR0 = wmma_bf16(al, nh0, accR0);
        accI0 = wmma_bf16(ah, krh0, accI0);
        accI0 = wmma_bf16(ah, krl0, accI0);
        accI0 = wmma_bf16(al, krh0, accI0);
        accR1 = wmma_bf16(ah, nh1, accR1);
        accR1 = wmma_bf16(ah, nl1, accR1);
        accR1 = wmma_bf16(al, nh1, accR1);
        accI1 = wmma_bf16(ah, krh1, accI1);
        accI1 = wmma_bf16(ah, krl1, accI1);
        accI1 = wmma_bf16(al, krh1, accI1);
      }
    }
  }

  // ---- epilogue: C/D layout M = v + 8*half, N = lane&15 ----
  const int c0 = nt0 * 16 + (lane & 15);
  const int c1 = nt1 * 16 + (lane & 15);
  const float bR0 = br[c0], bR1 = br[c1];
  const float bI0 = bi[c0], bI1 = bi[c1];
  const size_t rowbase = ((size_t)b * CH + h) * CW + w0 + m0;
  const size_t IMAG = (size_t)CB * CH * CW * CF;   // 33,554,432

#pragma unroll
  for (int v = 0; v < 8; ++v) {
    const int M = v + 8 * half;
    const size_t o = (rowbase + M) * CF;
    out[o + c0]        = accR0[v] + bR0;
    out[o + c1]        = accR1[v] + bR1;
    out[IMAG + o + c0] = accI0[v] + bI0;
    out[IMAG + o + c1] = accI1[v] + bI1;
  }
}

// ---------------------------------------------------------------------------
extern "C" void kernel_launch(void* const* d_in, const int* in_sizes, int n_in,
                              void* d_out, int out_size, void* d_ws, size_t ws_size,
                              hipStream_t stream) {
  const float* xr = (const float*)d_in[0];
  const float* xi = (const float*)d_in[1];
  const float* kr = (const float*)d_in[2];
  const float* ki = (const float*)d_in[3];
  const float* br = (const float*)d_in[4];
  const float* bi = (const float*)d_in[5];
  unsigned short* wsb = (unsigned short*)d_ws;   // bf16 hi/lo packed weights
  float* out = (float*)d_out;

  // 1) pack weights into WMMA B-fragment layout (runs once per launch)
  pack_weights_kernel<<<(WS_TOTAL + 255) / 256, 256, 0, stream>>>(kr, ki, wsb);

  // 2) main conv: grid = (W/32, H, B), 8 waves/block
  dim3 grid(CW / 32, CH, CB);
  complex_conv_wmma_kernel<<<grid, 256, 0, stream>>>(xr, xi, br, bi, wsb, out);
}
